// HDNet_91190745629219
// MI455X (gfx1250) — matrix operand order repaired and checked by reading, not compiled, not run on hardware
//
#include <hip/hip_runtime.h>

// ---------------------------------------------------------------------------
// Problem constants (from reference): B=4096, C=1024 -> GEMMs are 4096x2048x2048
// ---------------------------------------------------------------------------
#define BSZ 4096
#define CCH 1024
#define NN  2048   // 2C (output cols of each GEMM)
#define KK  2048   // 2C (inner dim of each GEMM)

typedef __attribute__((ext_vector_type(16))) _Float16 v16h;
typedef __attribute__((ext_vector_type(8)))  _Float16 v8h;
typedef __attribute__((ext_vector_type(4)))  _Float16 v4h;
typedef __attribute__((ext_vector_type(8)))  float    v8f;
typedef __attribute__((ext_vector_type(4)))  float    v4f;

// ---------------------------------------------------------------------------
// CDNA5 async global->LDS copies (ASYNCcnt path; bypasses VGPRs entirely).
// One fused asm block (no memory clobber): fewer scheduling walls so the
// register allocator can keep WMMA operand tuples contiguous.
// ---------------------------------------------------------------------------
__device__ __forceinline__ void async_stage4(unsigned la0, unsigned la1,
                                             unsigned lb0, unsigned lb1,
                                             const void* a0, const void* a1,
                                             const void* b0, const void* b1) {
  asm volatile(
      "global_load_async_to_lds_b128 %0, %4, off\n\t"
      "global_load_async_to_lds_b128 %1, %5, off\n\t"
      "global_load_async_to_lds_b128 %2, %6, off\n\t"
      "global_load_async_to_lds_b128 %3, %7, off"
      :: "v"(la0), "v"(la1), "v"(lb0), "v"(lb1),
         "v"(a0), "v"(a1), "v"(b0), "v"(b1));
}
__device__ __forceinline__ void wait_asynccnt0() {
  asm volatile("s_wait_asynccnt 0x0" ::: "memory");
}

// ---------------------------------------------------------------------------
// Kernel 1: concat(x0, x1) along dim1 and convert fp32 -> fp16 into A0 [B, 2C]
// ---------------------------------------------------------------------------
__global__ __launch_bounds__(256) void concat_convert_f16(
    const float* __restrict__ x0, const float* __restrict__ x1,
    _Float16* __restrict__ A)
{
  size_t idx = ((size_t)blockIdx.x * blockDim.x + threadIdx.x) * 4;
  int row = (int)(idx / NN);
  int col = (int)(idx % NN);
  const float* src = (col < CCH) ? (x0 + (size_t)row * CCH + col)
                                 : (x1 + (size_t)row * CCH + (col - CCH));
  v4f v = *(const v4f*)src;
  v4h h;
  h[0] = (_Float16)v[0]; h[1] = (_Float16)v[1];
  h[2] = (_Float16)v[2]; h[3] = (_Float16)v[3];
  *(v4h*)(A + idx) = h;
}

// ---------------------------------------------------------------------------
// Kernel 2: Wt[n][k] = (fp16) W[k][n]  (tiled transpose, one-time per weight)
// ---------------------------------------------------------------------------
__global__ __launch_bounds__(256) void transpose_w_f16(
    const float* __restrict__ W, _Float16* __restrict__ Wt)
{
  __shared__ float tile[32][33];
  const int tx = threadIdx.x & 31;
  const int ty = threadIdx.x >> 5;     // 0..7
  const int n0 = blockIdx.x * 32;
  const int k0 = blockIdx.y * 32;
  #pragma unroll
  for (int i = 0; i < 4; ++i) {
    int k = ty + i * 8;
    tile[k][tx] = W[(size_t)(k0 + k) * NN + n0 + tx];
  }
  __syncthreads();
  #pragma unroll
  for (int i = 0; i < 4; ++i) {
    int n = ty + i * 8;
    Wt[(size_t)(n0 + n) * KK + k0 + tx] = (_Float16)tile[tx][n];
  }
}

// ---------------------------------------------------------------------------
// Kernel 3: GEMM  D[M,N] = A[M,K] * Wt[N,K]^T   (fp16 in, fp32 accumulate)
//   - 256 threads = 8 waves, workgroup tile 128x128, BK=32
//   - LDS double buffer filled by global_load_async_to_lds_b128 (ASYNCcnt)
//   - branchless main loop (last iteration peeled) to keep accumulators pinned
//   - wave grid 4x2; each wave: 2 (M) x 4 (N) = 8 WMMA accumulators
// ---------------------------------------------------------------------------
constexpr int BM = 128, BN = 128, BK = 32;
constexpr int LDK = BK + 8;            // pad LDS rows to 40 halves (80 B)
constexpr int NKSTEPS = KK / BK;       // 64
constexpr unsigned BUFSTRIDE = BM * LDK * sizeof(_Float16);   // 10240 B
constexpr unsigned HALFTILE  = 64 * LDK * sizeof(_Float16);   // 5120 B

template <bool OUT_F32>
__global__ __launch_bounds__(256, 1) void gemm_f16_wmma(
    const _Float16* __restrict__ A,    // [M, KK] row-major fp16
    const _Float16* __restrict__ Wt,   // [NN, KK] row-major fp16 (W transposed)
    float* __restrict__ outF,          // used if OUT_F32
    _Float16* __restrict__ outH)       // used if !OUT_F32
{
  __shared__ __align__(16) _Float16 sA[2][BM][LDK];
  __shared__ __align__(16) _Float16 sB[2][BN][LDK];

  const int tid  = threadIdx.x;
  const int lane = tid & 31;
  const int wid  = tid >> 5;
  const int wm   = wid >> 1;           // 0..3 -> M offset wm*32
  const int wn   = wid & 1;            // 0..1 -> N offset wn*64
  const int m0   = blockIdx.y * BM;
  const int n0   = blockIdx.x * BN;

  // global->LDS staging: each thread moves two 16B chunks of A and of B
  const int r0 = tid >> 2;             // 0..63 (second chunk at r0+64)
  const int c0 = (tid & 3) * 8;        // 0,8,16,24 (halves within BK)

  // Precomputed bases: LDS byte offsets (buffer 0) and global row pointers
  const unsigned ldsA = (unsigned)(uintptr_t)&sA[0][r0][c0];
  const unsigned ldsB = (unsigned)(uintptr_t)&sB[0][r0][c0];
  const _Float16* apBase = A  + (size_t)(m0 + r0) * KK + c0;
  const _Float16* bpBase = Wt + (size_t)(n0 + r0) * KK + c0;

  auto stage = [&](int kt, int buf) {
    const _Float16* ap = apBase + (size_t)kt * BK;
    const _Float16* bp = bpBase + (size_t)kt * BK;
    unsigned la = ldsA + (unsigned)buf * BUFSTRIDE;
    unsigned lb = ldsB + (unsigned)buf * BUFSTRIDE;
    async_stage4(la, la + HALFTILE, lb, lb + HALFTILE,
                 ap, ap + (size_t)64 * KK, bp, bp + (size_t)64 * KK);
  };

  v8f acc[2][4];
  #pragma unroll
  for (int i = 0; i < 2; ++i)
    #pragma unroll
    for (int j = 0; j < 4; ++j)
      acc[i][j] = v8f{0.f, 0.f, 0.f, 0.f, 0.f, 0.f, 0.f, 0.f};

  // Per-lane fragment addressing (ISA 16-bit A 16x32 / B 32x16 layouts):
  const int lr   = lane & 15;
  const int hsel = lane >> 4;          // half-wave select
  const int akb  = hsel * 8;           // A: K in {akb..akb+7} u {akb+16..akb+23}
  const int bkb  = hsel * 8;           // B: K in {bkb..bkb+7} u {bkb+16..bkb+23}

  auto compute = [&](int buf) {
    union Frag { v16h v; v8h h[2]; };
    Frag a[2], b[4];
    #pragma unroll
    for (int i = 0; i < 2; ++i) {
      const _Float16* p = &sA[buf][wm * 32 + i * 16 + lr][akb];
      a[i].h[0] = *(const v8h*)p;
      a[i].h[1] = *(const v8h*)(p + 16);
    }
    #pragma unroll
    for (int j = 0; j < 4; ++j) {
      const _Float16* p = &sB[buf][wn * 64 + j * 16 + lr][bkb];
      b[j].h[0] = *(const v8h*)p;
      b[j].h[1] = *(const v8h*)(p + 16);
    }
    #pragma unroll
    for (int i = 0; i < 2; ++i)
      #pragma unroll
      for (int j = 0; j < 4; ++j)
        acc[i][j] = __builtin_amdgcn_wmma_f32_16x16x32_f16(
            false, a[i].v, false, b[j].v, (short)0, acc[i][j], false, false);
  };

  stage(0, 0);
  wait_asynccnt0();
  __syncthreads();

  // Branchless steady state: stage kt+1 while computing kt
  for (int kt = 0; kt < NKSTEPS - 1; ++kt) {
    stage(kt + 1, (kt + 1) & 1);
    compute(kt & 1);
    wait_asynccnt0();   // next buffer resident in LDS before the barrier
    __syncthreads();
  }
  compute((NKSTEPS - 1) & 1);   // peeled final step (nothing left to stage)

  // Epilogue: C/D layout -> VGPR v holds row (v + hsel*8), lane&15 is column
  const int rbase = m0 + wm * 32 + hsel * 8;
  const int cbase = n0 + wn * 64 + lr;
  #pragma unroll
  for (int i = 0; i < 2; ++i)
    #pragma unroll
    for (int j = 0; j < 4; ++j)
      #pragma unroll
      for (int v = 0; v < 8; ++v) {
        size_t off = (size_t)(rbase + i * 16 + v) * NN + (cbase + j * 16);
        if (OUT_F32) outF[off] = acc[i][j][v];
        else         outH[off] = (_Float16)acc[i][j][v];
      }
}

// ---------------------------------------------------------------------------
// Launch: convert + transpose once, then 3 chained WMMA GEMMs.
// Workspace layout (56 MB): Act0(16M) Act1(16M) Wt0(8M) Wt1(8M) Wt2(8M)
// ---------------------------------------------------------------------------
extern "C" void kernel_launch(void* const* d_in, const int* in_sizes, int n_in,
                              void* d_out, int out_size, void* d_ws, size_t ws_size,
                              hipStream_t stream)
{
  const float* x0 = (const float*)d_in[0];
  const float* x1 = (const float*)d_in[1];
  const float* W0 = (const float*)d_in[2];
  const float* W1 = (const float*)d_in[3];
  const float* W2 = (const float*)d_in[4];

  char* ws = (char*)d_ws;
  _Float16* Act0 = (_Float16*)(ws);
  _Float16* Act1 = (_Float16*)(ws + ((size_t)16 << 20));
  _Float16* Wt0  = (_Float16*)(ws + ((size_t)32 << 20));
  _Float16* Wt1  = (_Float16*)(ws + ((size_t)40 << 20));
  _Float16* Wt2  = (_Float16*)(ws + ((size_t)48 << 20));

  // Stage inputs to fp16
  concat_convert_f16<<<((size_t)BSZ * NN / 4) / 256, 256, 0, stream>>>(x0, x1, Act0);
  dim3 tg(NN / 32, KK / 32);
  transpose_w_f16<<<tg, 256, 0, stream>>>(W0, Wt0);
  transpose_w_f16<<<tg, 256, 0, stream>>>(W1, Wt1);
  transpose_w_f16<<<tg, 256, 0, stream>>>(W2, Wt2);

  // Three chained GEMMs (kernel boundaries = global barriers between layers)
  dim3 gg(NN / BN, BSZ / BM);  // (16, 32)
  gemm_f16_wmma<false><<<gg, 256, 0, stream>>>(Act0, Wt0, nullptr, Act1);
  gemm_f16_wmma<false><<<gg, 256, 0, stream>>>(Act1, Wt1, nullptr, Act0);
  gemm_f16_wmma<true ><<<gg, 256, 0, stream>>>(Act0, Wt2, (float*)d_out, nullptr);
}
